// BGConv_unit_78340203479084
// MI455X (gfx1250) — matrix operand order repaired and checked by reading, not compiled, not run on hardware
//
#include <hip/hip_runtime.h>
#include <hip/hip_bf16.h>

// ---------------------------------------------------------------------------
// BGConv fused kernel for gfx1250 (MI455X), wave32 + WMMA f16->f32.
//   feats: [N_NODES,256] f32   pairs: [N_EDGES,2] i32   conf: [N_EDGES] f32
//   W1:[512,512] b1:[512] W2:[512,512] b2:[512]
//   out = segment-softmax aggregation of 2-layer MLP over edge-concat feats
// ---------------------------------------------------------------------------

typedef _Float16 v8h  __attribute__((ext_vector_type(8)));
typedef _Float16 v16h __attribute__((ext_vector_type(16)));
typedef float    v8f  __attribute__((ext_vector_type(8)));

#define DD     256
#define TWO_D  512
#define HID    512
#define MT     32            // edges per block
#define AROW   520           // padded LDS row stride (halves): 1040B, +4 banks/row
#define CONSTM 10.0f

__device__ __forceinline__ unsigned fenc(float f) {
  unsigned u = __float_as_uint(f);
  return (u & 0x80000000u) ? ~u : (u | 0x80000000u);
}
__device__ __forceinline__ float fdec(unsigned u) {
  return __uint_as_float((u & 0x80000000u) ? (u & 0x7fffffffu) : ~u);
}

__device__ __forceinline__ v8f wmma16(v16h a, v16h b, v8f c) {
  // v_wmma_f32_16x16x32_f16: (neg_a, A, neg_b, B, c_mod, C, reuse_a, reuse_b)
  return __builtin_amdgcn_wmma_f32_16x16x32_f16(false, a, false, b, (short)0, c,
                                                false, false);
}

// --- weight prep: fp32 [K][N] -> f16 transposed [N][K] ----------------------
__global__ void k_prep(const float* __restrict__ W1, const float* __restrict__ W2,
                       _Float16* __restrict__ W1t, _Float16* __restrict__ W2t) {
  int idx  = blockIdx.x * blockDim.x + threadIdx.x;   // 0 .. 2*512*512-1
  int half = idx >> 18;
  int r    = idx & 262143;
  int n    = r >> 9;
  int k    = r & 511;
  if (half == 0) W1t[(size_t)n * 512 + k] = (_Float16)W1[(size_t)k * 512 + n];
  else           W2t[(size_t)n * 512 + k] = (_Float16)W2[(size_t)k * 512 + n];
}

// --- segment max of confidence (exact, order-encoded uints) ----------------
__global__ void k_init_max(unsigned* __restrict__ mEnc, int nNodes) {
  int i = blockIdx.x * blockDim.x + threadIdx.x;
  if (i < nNodes) mEnc[i] = fenc(CONSTM);
}

__global__ void k_edge_max(const int* __restrict__ pairs,
                           const float* __restrict__ conf,
                           unsigned* __restrict__ mEnc, int nEdges) {
  int e = blockIdx.x * blockDim.x + threadIdx.x;
  if (e < nEdges) {
    unsigned c = fenc(conf[e]);
    atomicMax(&mEnc[pairs[2 * e]],     c);
    atomicMax(&mEnc[pairs[2 * e + 1]], c);
  }
}

// --- decode m, seed denom + numer(self term) into d_out --------------------
__global__ void k_node_finish(const unsigned* __restrict__ mEnc,
                              const float* __restrict__ feats,
                              float* __restrict__ mArr,
                              float* __restrict__ denom,
                              float* __restrict__ numer, int nNodes) {
  int idx = blockIdx.x * blockDim.x + threadIdx.x;
  int n = idx >> 8, d = idx & 255;
  if (n < nNodes) {
    float m  = fdec(mEnc[n]);
    float sw = __expf(CONSTM - m);
    if (d == 0) { mArr[n] = m; denom[n] = sw; }
    numer[idx] = sw * feats[idx];
  }
}

// --- fused gather + MLP (WMMA) + weighted scatter --------------------------
__global__ __launch_bounds__(256)
void k_mlp(const float* __restrict__ feats,
           const int*   __restrict__ pairs,
           const float* __restrict__ conf,
           const float* __restrict__ b1,
           const float* __restrict__ b2,
           const _Float16* __restrict__ W1t,
           const _Float16* __restrict__ W2t,
           const float* __restrict__ mArr,
           float* __restrict__ denom,
           float* __restrict__ numer,
           int nEdges) {
  __shared__ _Float16 As[MT * AROW];   // 33280 B, edge-concat feats, f16
  __shared__ _Float16 Hs[MT * AROW];   // 33280 B, hidden activations, f16
  __shared__ int   sSub[MT], sObj[MT];
  __shared__ float sWs[MT],  sWo[MT];

  const int tid = threadIdx.x;
  const int e0  = blockIdx.x * MT;

  // per-edge metadata + denom contribution (once per edge)
  if (tid < MT) {
    int e = e0 + tid;
    int s = 0, o = 0; float ws = 0.f, wo = 0.f;
    if (e < nEdges) {
      s = pairs[2 * e]; o = pairs[2 * e + 1];
      float c = conf[e];
      ws = __expf(c - mArr[s]);
      wo = __expf(c - mArr[o]);
      __hip_atomic_fetch_add(&denom[s], ws, __ATOMIC_RELAXED, __HIP_MEMORY_SCOPE_AGENT);
      __hip_atomic_fetch_add(&denom[o], wo, __ATOMIC_RELAXED, __HIP_MEMORY_SCOPE_AGENT);
    }
    sSub[tid] = s; sObj[tid] = o; sWs[tid] = ws; sWo[tid] = wo;
  }

  // gather: each thread fills 64 contiguous cols of one edge-row (f32->f16)
  {
    int row = tid >> 3, seg = tid & 7;
    int e = e0 + row;
    _Float16* dst = &As[row * AROW + seg * 64];
    if (e < nEdges) {
      int colBase = seg * 64;
      int node = (colBase < DD) ? pairs[2 * e] : pairs[2 * e + 1];
      const float4* src = (const float4*)(feats + (size_t)node * DD + (colBase & (DD - 1)));
      #pragma unroll
      for (int k = 0; k < 8; ++k) {
        float4 a = src[2 * k], b = src[2 * k + 1];
        v8h h;
        h[0] = (_Float16)a.x; h[1] = (_Float16)a.y; h[2] = (_Float16)a.z; h[3] = (_Float16)a.w;
        h[4] = (_Float16)b.x; h[5] = (_Float16)b.y; h[6] = (_Float16)b.z; h[7] = (_Float16)b.w;
        *(v8h*)(dst + 8 * k) = h;
      }
    } else {
      v8h z = {};
      #pragma unroll
      for (int k = 0; k < 8; ++k) *(v8h*)(dst + 8 * k) = z;
    }
  }
  __syncthreads();

  const int lane = tid & 31;
  const int wv   = tid >> 5;     // wave 0..7
  const int lh   = lane >> 4;    // lane half
  const int lm   = lane & 15;

  v8f acc[2][4];

  // ---------------- layer 1: A(32x512) @ W1t -> H, ReLU ----------------
  #pragma unroll
  for (int j = 0; j < 4; ++j) {
    float bb = b1[(wv * 4 + j) * 16 + lm];
    #pragma unroll
    for (int mi = 0; mi < 2; ++mi)
      #pragma unroll
      for (int r = 0; r < 8; ++r) acc[mi][j][r] = bb;
  }
  for (int s = 0; s < TWO_D / 32; ++s) {
    // A frags: lane holds M=lm; K-chunks {32s+8*lh .. +8} and {32s+16+8*lh .. +8}
    const _Float16* ap0 = &As[lm * AROW + 32 * s + 8 * lh];
    v8h lo = *(const v8h*)ap0;
    v8h hi = *(const v8h*)(ap0 + 16);
    v16h a0 = __builtin_shufflevector(lo, hi, 0,1,2,3,4,5,6,7,8,9,10,11,12,13,14,15);
    const _Float16* ap1 = ap0 + 16 * AROW;
    lo = *(const v8h*)ap1;
    hi = *(const v8h*)(ap1 + 16);
    v16h a1 = __builtin_shufflevector(lo, hi, 0,1,2,3,4,5,6,7,8,9,10,11,12,13,14,15);
    #pragma unroll
    for (int j = 0; j < 4; ++j) {
      int n = (wv * 4 + j) * 16 + lm;
      // B frag: lane holds N=lm(+tile); K-chunks {32s+8*lh .. +8} and {32s+16+8*lh .. +8}
      const _Float16* bp = W1t + (size_t)n * TWO_D + 32 * s + 8 * lh;
      v16h b = __builtin_shufflevector(*(const v8h*)bp, *(const v8h*)(bp + 16), 0,1,2,3,4,5,6,7,8,9,10,11,12,13,14,15);
      acc[0][j] = wmma16(a0, b, acc[0][j]);
      acc[1][j] = wmma16(a1, b, acc[1][j]);
    }
  }
  // ReLU -> Hs (C/D layout: VGPR r <-> row r + 8*lh within 16-tile)
  #pragma unroll
  for (int j = 0; j < 4; ++j) {
    int n = (wv * 4 + j) * 16 + lm;
    #pragma unroll
    for (int mi = 0; mi < 2; ++mi)
      #pragma unroll
      for (int r = 0; r < 8; ++r) {
        int m = mi * 16 + r + 8 * lh;
        float v = acc[mi][j][r];
        Hs[m * AROW + n] = (_Float16)(v > 0.f ? v : 0.f);
      }
  }
  __syncthreads();

  // ---------------- layer 2: H(32x512) @ W2t + b2 ----------------
  #pragma unroll
  for (int j = 0; j < 4; ++j) {
    float bb = b2[(wv * 4 + j) * 16 + lm];
    #pragma unroll
    for (int mi = 0; mi < 2; ++mi)
      #pragma unroll
      for (int r = 0; r < 8; ++r) acc[mi][j][r] = bb;
  }
  for (int s = 0; s < HID / 32; ++s) {
    const _Float16* ap0 = &Hs[lm * AROW + 32 * s + 8 * lh];
    v8h lo = *(const v8h*)ap0;
    v8h hi = *(const v8h*)(ap0 + 16);
    v16h a0 = __builtin_shufflevector(lo, hi, 0,1,2,3,4,5,6,7,8,9,10,11,12,13,14,15);
    const _Float16* ap1 = ap0 + 16 * AROW;
    lo = *(const v8h*)ap1;
    hi = *(const v8h*)(ap1 + 16);
    v16h a1 = __builtin_shufflevector(lo, hi, 0,1,2,3,4,5,6,7,8,9,10,11,12,13,14,15);
    #pragma unroll
    for (int j = 0; j < 4; ++j) {
      int n = (wv * 4 + j) * 16 + lm;
      const _Float16* bp = W2t + (size_t)n * HID + 32 * s + 8 * lh;
      v16h b = __builtin_shufflevector(*(const v8h*)bp, *(const v8h*)(bp + 16), 0,1,2,3,4,5,6,7,8,9,10,11,12,13,14,15);
      acc[0][j] = wmma16(a0, b, acc[0][j]);
      acc[1][j] = wmma16(a1, b, acc[1][j]);
    }
  }

  // ---------------- weighted scatter into numer (d_out) ----------------
  #pragma unroll
  for (int j = 0; j < 4; ++j) {
    int c = (wv * 4 + j) * 16 + lm;        // 0..511
    bool isSub = (c < DD);
    int dcol = isSub ? c : c - DD;
    #pragma unroll
    for (int mi = 0; mi < 2; ++mi)
      #pragma unroll
      for (int r = 0; r < 8; ++r) {
        int m = mi * 16 + r + 8 * lh;
        int e = e0 + m;
        if (e < nEdges) {
          int   node = isSub ? sSub[m] : sObj[m];
          float w    = isSub ? sWs[m]  : sWo[m];
          __hip_atomic_fetch_add(&numer[(size_t)node * DD + dcol],
                                 w * acc[mi][j][r],
                                 __ATOMIC_RELAXED, __HIP_MEMORY_SCOPE_AGENT);
        }
      }
  }
}

// --- numer / denom ----------------------------------------------------------
__global__ void k_final(float* __restrict__ out, const float* __restrict__ denom,
                        int nNodes) {
  int idx = blockIdx.x * blockDim.x + threadIdx.x;
  int n = idx >> 8;
  if (n < nNodes) out[idx] = out[idx] / denom[n];
}

extern "C" void kernel_launch(void* const* d_in, const int* in_sizes, int n_in,
                              void* d_out, int out_size, void* d_ws, size_t ws_size,
                              hipStream_t stream) {
  const float* feats = (const float*)d_in[0];
  const int*   pairs = (const int*)d_in[1];
  const float* conf  = (const float*)d_in[2];
  const float* W1    = (const float*)d_in[3];
  const float* b1    = (const float*)d_in[4];
  const float* W2    = (const float*)d_in[5];
  const float* b2    = (const float*)d_in[6];
  float* out = (float*)d_out;

  const int nNodes = in_sizes[0] / DD;
  const int nEdges = in_sizes[2];

  // workspace layout (256B aligned slices)
  char* w = (char*)d_ws;
  size_t off = 0;
  auto take = [&](size_t bytes) {
    void* p = w + off;
    off += (bytes + 255) & ~(size_t)255;
    return p;
  };
  _Float16* W1t   = (_Float16*)take((size_t)HID * TWO_D * sizeof(_Float16));
  _Float16* W2t   = (_Float16*)take((size_t)TWO_D * HID * sizeof(_Float16));
  unsigned* mEnc  = (unsigned*)take((size_t)nNodes * sizeof(unsigned));
  float*    mArr  = (float*)take((size_t)nNodes * sizeof(float));
  float*    denom = (float*)take((size_t)nNodes * sizeof(float));
  (void)ws_size;

  // 1) f16 transposed weights
  k_prep<<<(2 * 512 * 512) / 256, 256, 0, stream>>>(W1, W2, W1t, W2t);

  // 2) segment max of confidence
  k_init_max<<<(nNodes + 255) / 256, 256, 0, stream>>>(mEnc, nNodes);
  k_edge_max<<<(nEdges + 255) / 256, 256, 0, stream>>>(pairs, conf, mEnc, nEdges);

  // 3) decode m, seed denom + self-term numer in d_out
  {
    size_t total = (size_t)nNodes * DD;
    k_node_finish<<<(unsigned)((total + 255) / 256), 256, 0, stream>>>(
        mEnc, feats, mArr, denom, out, nNodes);
  }

  // 4) fused gather + MLP (WMMA) + weighted scatter
  k_mlp<<<(nEdges + MT - 1) / MT, 256, 0, stream>>>(
      feats, pairs, conf, b1, b2, W1t, W2t, mArr, denom, out, nEdges);

  // 5) normalize
  {
    size_t total = (size_t)nNodes * DD;
    k_final<<<(unsigned)((total + 255) / 256), 256, 0, stream>>>(out, denom, nNodes);
  }
}
